// OTAlignmentLayer_59940563583135
// MI455X (gfx1250) — matrix operand order repaired and checked by reading, not compiled, not run on hardware
//
#include <hip/hip_runtime.h>
#include <hip/hip_bf16.h>
#include <math.h>

// ---------------------------------------------------------------------------
// OT alignment layer for MI455X (gfx1250, wave32, WMMA).
// GEMMs: v_wmma_f32_16x16x32_bf16, B-operand staged in double-buffered LDS
// (cooperative per-block load, 8x less L2 traffic, ds-latency fragment reads),
// A-operand software-pipelined in registers.
// Sinkhorn stays fp32 (bandwidth-bound; log_K = 134MB fits MI455X's 192MB L2).
// ---------------------------------------------------------------------------

typedef __bf16 bf16;
typedef __attribute__((ext_vector_type(16))) __bf16 v16bf;
typedef __attribute__((ext_vector_type(8)))  float  v8f;

static constexpr int BB = 16;    // batch
static constexpr int LT = 1024;  // text length
static constexpr int LA = 2048;  // audio length
static constexpr int DD = 1024;  // feature dim

static constexpr int BROW = 64;  // B tile rows (N per block)
static constexpr int BPAD = 40;  // padded row stride in bf16 (80B -> bank-spread)

#define SINK_EPS_F 0.1f
#define SINK_ITERS 10
#define BIGF 1.0e9f

__device__ inline bf16 f2bf(float x) {
  unsigned u = __builtin_bit_cast(unsigned, x);
  u += 0x7FFFu + ((u >> 16) & 1u);             // round-to-nearest-even
  unsigned short h = (unsigned short)(u >> 16);
  return __builtin_bit_cast(bf16, h);
}

__device__ inline float cleanf(float x) {
  if (x != x) return 0.0f;                      // NaN -> 0
  return fminf(fmaxf(x, -BIGF), BIGF);
}

__device__ inline float block_reduce_sum256(float v) {
  __shared__ float red[256];
  int t = threadIdx.x;
  red[t] = v; __syncthreads();
  #pragma unroll
  for (int off = 128; off > 0; off >>= 1) {
    if (t < off) red[t] += red[t + off];
    __syncthreads();
  }
  float r = red[0];
  __syncthreads();
  return r;
}

// ------------------------- WMMA fragment helpers ---------------------------
// A-matrix 16x32 bf16 (ISA 7.12.2): lanes 0-15 M=lane, K 0..7 & 16..23;
// lanes 16-31 same M, K 8..15 & 24..31.
__device__ inline v16bf load_frag_a_bf16(const bf16* __restrict__ p, int ld) {
  const int lane = threadIdx.x & 31;
  const bf16* r = p + (size_t)(lane & 15) * ld + ((lane >> 4) << 3);
  v16bf f;
  #pragma unroll
  for (int i = 0; i < 8; ++i) f[i] = r[i];
  #pragma unroll
  for (int i = 0; i < 8; ++i) f[8 + i] = r[16 + i];
  return f;
}

// Cooperative stage of a 64(N) x 32(K) bf16 tile into LDS (padded rows).
// 256 threads x 16B each. src points at (n0, k0) of a row-major [N][K] buffer.
__device__ inline void stage_b(bf16* __restrict__ sB, const bf16* __restrict__ src,
                               int ld) {
  const int t = threadIdx.x;
  const int row = t >> 2;
  const int part = (t & 3) << 3;                // 0,8,16,24 (bf16 elems)
  const uint4 v = *(const uint4*)(src + (size_t)row * ld + part);
  *(uint4*)(sB + row * BPAD + part) = v;
}

// B-matrix 32x16 fragment from the LDS tile: lane n = lane&15 (+nrow),
// lanes 0-15 K 0..7 & 16..23; lanes 16-31 K 8..15 & 24..31 (matches A).
__device__ inline v16bf load_frag_b_lds(const bf16* __restrict__ sB, int nrow) {
  const int lane = threadIdx.x & 31;
  const bf16* r = sB + (nrow + (lane & 15)) * BPAD + ((lane >> 4) << 3);
  v16bf f;
  #pragma unroll
  for (int i = 0; i < 8; ++i) f[i] = r[i];
  #pragma unroll
  for (int i = 0; i < 8; ++i) f[8 + i] = r[16 + i];
  return f;
}

// A-fragment of pi computed on the fly: pi = exp(u + logK + v), cast to bf16.
__device__ inline v16bf load_frag_pi(const float* __restrict__ krow,
                                     const float* __restrict__ vrow, float u) {
  const int g8 = ((threadIdx.x & 31) >> 4) << 3;
  v16bf f;
  #pragma unroll
  for (int i = 0; i < 8; ++i) {
    int k = g8 + i;
    f[i] = f2bf(__expf(u + krow[k] + vrow[k]));
  }
  #pragma unroll
  for (int i = 0; i < 8; ++i) {
    int k = 16 + g8 + i;
    f[8 + i] = f2bf(__expf(u + krow[k] + vrow[k]));
  }
  return f;
}

// --------------------------- elementwise kernels ---------------------------
__global__ __launch_bounds__(256)
void cast_bf16_kernel(const float* __restrict__ x, bf16* __restrict__ y, int n) {
  for (int i = blockIdx.x * 256 + threadIdx.x; i < n; i += gridDim.x * 256)
    y[i] = f2bf(x[i]);
}

__global__ __launch_bounds__(256)
void zero_kernel(float* __restrict__ p, int n) {
  for (int i = blockIdx.x * 256 + threadIdx.x; i < n; i += gridDim.x * 256)
    p[i] = 0.0f;
}

// LayerNorm row -> bf16
__global__ __launch_bounds__(256)
void ln_cast_kernel(const float* __restrict__ x, const float* __restrict__ gam,
                    const float* __restrict__ bet, bf16* __restrict__ out) {
  const int row = blockIdx.x;
  const float* xr = x + (size_t)row * DD;
  float s = 0.0f;
  for (int i = threadIdx.x; i < DD; i += 256) s += xr[i];
  const float mu = block_reduce_sum256(s) * (1.0f / DD);
  float s2 = 0.0f;
  for (int i = threadIdx.x; i < DD; i += 256) { float t = xr[i] - mu; s2 += t * t; }
  const float var = block_reduce_sum256(s2) * (1.0f / DD);
  const float rstd = rsqrtf(var + 1e-5f);
  bf16* orow = out + (size_t)row * DD;
  for (int i = threadIdx.x; i < DD; i += 256)
    orow[i] = f2bf((xr[i] - mu) * rstd * gam[i] + bet[i]);
}

// L2-normalize row (fp32 in) -> bf16 out
__global__ __launch_bounds__(256)
void l2cast_kernel(const float* __restrict__ Y, bf16* __restrict__ out) {
  const int row = blockIdx.x;
  const float* yr = Y + (size_t)row * DD;
  float s = 0.0f;
  for (int i = threadIdx.x; i < DD; i += 256) s += yr[i] * yr[i];
  const float nrm = sqrtf(block_reduce_sum256(s));
  const float inv = 1.0f / fmaxf(nrm, 1e-12f);
  bf16* orow = out + (size_t)row * DD;
  for (int i = threadIdx.x; i < DD; i += 256) orow[i] = f2bf(yr[i] * inv);
}

// log_mu / log_nu from int mask
__global__ __launch_bounds__(256)
void mask_log_kernel(const int* __restrict__ mask, float* __restrict__ out, int L) {
  const int b = blockIdx.x;
  const int* mr = mask + (size_t)b * L;
  float s = 0.0f;
  for (int i = threadIdx.x; i < L; i += 256) s += (float)mr[i];
  const float len = fmaxf(block_reduce_sum256(s), 1.0f);
  for (int i = threadIdx.x; i < L; i += 256)
    out[(size_t)b * L + i] = __logf((float)mr[i] / len + 1e-20f);
}

// ------------------------------ WMMA GEMMs ---------------------------------
// Projection: C[m,n] = sum_k A[m,k] * Wt[n,k] + bias[n]   (rows = B*L flattened)
__global__ __launch_bounds__(256)
void gemm_ln_proj_kernel(const bf16* __restrict__ A, const bf16* __restrict__ Wt,
                         const float* __restrict__ bias, float* __restrict__ C,
                         int N, int K) {
  __shared__ bf16 sB[2][BROW * BPAD];
  const int wave = threadIdx.x >> 5;
  const int lane = threadIdx.x & 31;
  const int m0 = blockIdx.y * 128 + wave * 16;
  const int n0 = blockIdx.x * 64;
  const bf16* Ab = A + (size_t)m0 * K;
  const bf16* Bb = Wt + (size_t)n0 * K;

  stage_b(sB[0], Bb, K);
  v16bf af = load_frag_a_bf16(Ab, K);
  __syncthreads();

  v8f acc[4] = {};
  for (int k0 = 0; k0 < K; k0 += 32) {
    const int cur = (k0 >> 5) & 1;
    const bool more = (k0 + 32) < K;
    if (more) stage_b(sB[cur ^ 1], Bb + k0 + 32, K);
    v16bf a_now = af;
    if (more) {
      __builtin_prefetch(Ab + k0 + 1024, 0, 1);
      af = load_frag_a_bf16(Ab + k0 + 32, K);
    }
    #pragma unroll
    for (int j = 0; j < 4; ++j) {
      v16bf bfrag = load_frag_b_lds(sB[cur], j * 16);
      acc[j] = __builtin_amdgcn_wmma_f32_16x16x32_bf16(
          false, a_now, false, bfrag, (short)0, acc[j], false, false);
    }
    __syncthreads();
  }

  const int n = lane & 15, g = lane >> 4;
  #pragma unroll
  for (int j = 0; j < 4; ++j) {
    const int col = n0 + j * 16 + n;
    const float bv = bias[col];
    float* Cp = C + (size_t)(m0 + g * 8) * N + col;
    #pragma unroll
    for (int r = 0; r < 8; ++r) Cp[(size_t)r * N] = acc[j][r] + bv;
  }
}

// cost GEMM: log_K[b,m,n] = (dot(Tn[b,m,:],An[b,n,:]) - 1) / eps
__global__ __launch_bounds__(256)
void gemm_cost_kernel(const bf16* __restrict__ Tn, const bf16* __restrict__ An,
                      float* __restrict__ lK) {
  __shared__ bf16 sB[2][BROW * BPAD];
  const int b = blockIdx.z;
  const int wave = threadIdx.x >> 5;
  const int lane = threadIdx.x & 31;
  const int m0 = blockIdx.y * 128 + wave * 16;
  const int n0 = blockIdx.x * 64;
  const bf16* Ab = Tn + ((size_t)b * LT + m0) * DD;
  const bf16* Bb = An + ((size_t)b * LA + n0) * DD;

  stage_b(sB[0], Bb, DD);
  v16bf af = load_frag_a_bf16(Ab, DD);
  __syncthreads();

  v8f acc[4] = {};
  for (int k0 = 0; k0 < DD; k0 += 32) {
    const int cur = (k0 >> 5) & 1;
    const bool more = (k0 + 32) < DD;
    if (more) stage_b(sB[cur ^ 1], Bb + k0 + 32, DD);
    v16bf a_now = af;
    if (more) af = load_frag_a_bf16(Ab + k0 + 32, DD);
    #pragma unroll
    for (int j = 0; j < 4; ++j) {
      v16bf bfrag = load_frag_b_lds(sB[cur], j * 16);
      acc[j] = __builtin_amdgcn_wmma_f32_16x16x32_bf16(
          false, a_now, false, bfrag, (short)0, acc[j], false, false);
    }
    __syncthreads();
  }

  const float inv_eps = 1.0f / (SINK_EPS_F + 1e-8f);
  const int n = lane & 15, g = lane >> 4;
  #pragma unroll
  for (int j = 0; j < 4; ++j) {
    const int col = n0 + j * 16 + n;
    float* Kp = lK + ((size_t)b * LT + m0 + g * 8) * LA + col;
    #pragma unroll
    for (int r = 0; r < 8; ++r) Kp[(size_t)r * LA] = (acc[j][r] - 1.0f) * inv_eps;
  }
}

// final GEMM: A_aligned[b,m,c] = sum_n exp(u[m]+logK[m,n]+v[n]) * At[b,c,n]
__global__ __launch_bounds__(256)
void gemm_pi_align_kernel(const float* __restrict__ lK, const float* __restrict__ lu,
                          const float* __restrict__ lv, const bf16* __restrict__ At,
                          float* __restrict__ Aal) {
  __shared__ bf16 sB[2][BROW * BPAD];
  const int b = blockIdx.z;
  const int wave = threadIdx.x >> 5;
  const int lane = threadIdx.x & 31;
  const int m0 = blockIdx.y * 128 + wave * 16;
  const int n0 = blockIdx.x * 64;
  const float* Kb = lK + ((size_t)b * LT + m0 + (lane & 15)) * LA;
  const float* vb = lv + (size_t)b * LA;
  const float  u  = lu[(size_t)b * LT + m0 + (lane & 15)];
  const bf16*  Bb = At + ((size_t)b * DD + n0) * LA;

  stage_b(sB[0], Bb, LA);
  v16bf af = load_frag_pi(Kb, vb, u);
  __syncthreads();

  v8f acc[4] = {};
  for (int k0 = 0; k0 < LA; k0 += 32) {
    const int cur = (k0 >> 5) & 1;
    const bool more = (k0 + 32) < LA;
    if (more) stage_b(sB[cur ^ 1], Bb + k0 + 32, LA);
    v16bf a_now = af;
    if (more) af = load_frag_pi(Kb + k0 + 32, vb + k0 + 32, u);
    #pragma unroll
    for (int j = 0; j < 4; ++j) {
      v16bf bfrag = load_frag_b_lds(sB[cur], j * 16);
      acc[j] = __builtin_amdgcn_wmma_f32_16x16x32_bf16(
          false, a_now, false, bfrag, (short)0, acc[j], false, false);
    }
    __syncthreads();
  }

  const int n = lane & 15, g = lane >> 4;
  #pragma unroll
  for (int j = 0; j < 4; ++j) {
    const int col = n0 + j * 16 + n;
    float* Op = Aal + ((size_t)b * LT + m0 + g * 8) * DD + col;
    #pragma unroll
    for (int r = 0; r < 8; ++r) Op[(size_t)r * DD] = acc[j][r];
  }
}

// LDS-tiled transpose+cast: A_prime [b][La][d] f32 -> At [b][d][La] bf16
__global__ __launch_bounds__(256)
void transpose_cast_kernel(const float* __restrict__ X, bf16* __restrict__ Y) {
  __shared__ bf16 tile[32][33];
  const int b = blockIdx.z;
  const float* Xb = X + (size_t)b * LA * DD;
  bf16* Yb = Y + (size_t)b * DD * LA;
  const int c0 = blockIdx.x * 32;      // over d
  const int r0 = blockIdx.y * 32;      // over La
  const int tx = threadIdx.x & 31, ty = threadIdx.x >> 5;  // 32 x 8
  #pragma unroll
  for (int i = 0; i < 4; ++i) {
    int r = r0 + ty + i * 8;
    tile[ty + i * 8][tx] = f2bf(Xb[(size_t)r * DD + c0 + tx]);
  }
  __syncthreads();
  #pragma unroll
  for (int i = 0; i < 4; ++i) {
    int c = c0 + ty + i * 8;
    Yb[(size_t)c * LA + r0 + tx] = tile[tx][ty + i * 8];
  }
}

// --------------------------- Sinkhorn iterations ---------------------------
// log_u[b,m] = clean(log_mu - LSE_n(logK[b,m,n] + log_v[b,n]))
__global__ __launch_bounds__(256)
void sink_u_kernel(const float* __restrict__ lK, const float* __restrict__ lv,
                   const float* __restrict__ lmu, float* __restrict__ lu) {
  const int b = blockIdx.y, m = blockIdx.x;
  const float* Kr = lK + ((size_t)b * LT + m) * LA;
  const float* vr = lv + (size_t)b * LA;
  float mx = -__builtin_inff(), s = 0.0f;
  for (int n = threadIdx.x; n < LA; n += 256) {
    float t = Kr[n] + vr[n];
    if (t > mx) { s = s * __expf(mx - t) + 1.0f; mx = t; }
    else        { s += __expf(t - mx); }
  }
  __shared__ float smax[256], ssum[256];
  smax[threadIdx.x] = mx; ssum[threadIdx.x] = s;
  __syncthreads();
  for (int off = 128; off > 0; off >>= 1) {
    if (threadIdx.x < off) {
      float m1 = smax[threadIdx.x], s1 = ssum[threadIdx.x];
      float m2 = smax[threadIdx.x + off], s2 = ssum[threadIdx.x + off];
      float mm = fmaxf(m1, m2);
      float ss = (mm == -__builtin_inff())
                 ? 0.0f : s1 * __expf(m1 - mm) + s2 * __expf(m2 - mm);
      smax[threadIdx.x] = mm; ssum[threadIdx.x] = ss;
    }
    __syncthreads();
  }
  if (threadIdx.x == 0) {
    float lse = (ssum[0] > 0.0f) ? (smax[0] + __logf(ssum[0])) : -BIGF;
    lu[(size_t)b * LT + m] = cleanf(lmu[(size_t)b * LT + m] - lse);
  }
}

// log_v[b,n] = clean(log_nu - LSE_m(logK[b,m,n] + log_u[b,m]))
__global__ __launch_bounds__(256)
void sink_v_kernel(const float* __restrict__ lK, const float* __restrict__ lu,
                   const float* __restrict__ lnu, float* __restrict__ lv) {
  const int b = blockIdx.y;
  const int n = blockIdx.x * 64 + threadIdx.x;  // threadIdx.x in [0,64)
  const int tm = threadIdx.y;                   // [0,4)
  const float* Kb = lK + (size_t)b * LT * LA;
  const float* ub = lu + (size_t)b * LT;
  float mx = -__builtin_inff(), s = 0.0f;
  for (int m = tm; m < LT; m += 4) {
    float t = Kb[(size_t)m * LA + n] + ub[m];
    if (t > mx) { s = s * __expf(mx - t) + 1.0f; mx = t; }
    else        { s += __expf(t - mx); }
  }
  __shared__ float smax[4][64], ssum[4][64];
  smax[tm][threadIdx.x] = mx; ssum[tm][threadIdx.x] = s;
  __syncthreads();
  if (tm == 0) {
    float m1 = smax[0][threadIdx.x], s1 = ssum[0][threadIdx.x];
    #pragma unroll
    for (int j = 1; j < 4; ++j) {
      float m2 = smax[j][threadIdx.x], s2 = ssum[j][threadIdx.x];
      float mm = fmaxf(m1, m2);
      float ss = (mm == -__builtin_inff())
                 ? 0.0f : s1 * __expf(m1 - mm) + s2 * __expf(m2 - mm);
      m1 = mm; s1 = ss;
    }
    float lse = (s1 > 0.0f) ? (m1 + __logf(s1)) : -BIGF;
    lv[(size_t)b * LA + n] = cleanf(lnu[(size_t)b * LA + n] - lse);
  }
}

// S[b,m] = sum_d T_prime * A_aligned
__global__ __launch_bounds__(256)
void s_dot_kernel(const float* __restrict__ T, const float* __restrict__ A,
                  float* __restrict__ S) {
  const int row = blockIdx.x;
  const float* tr = T + (size_t)row * DD;
  const float* ar = A + (size_t)row * DD;
  float s = 0.0f;
  for (int i = threadIdx.x; i < DD; i += 256) s += tr[i] * ar[i];
  s = block_reduce_sum256(s);
  if (threadIdx.x == 0) S[row] = s;
}

// ------------------------------- launcher ----------------------------------
extern "C" void kernel_launch(void* const* d_in, const int* in_sizes, int n_in,
                              void* d_out, int out_size, void* d_ws, size_t ws_size,
                              hipStream_t stream) {
  (void)in_sizes; (void)n_in; (void)out_size; (void)ws_size;

  const float* text  = (const float*)d_in[0];
  const float* audio = (const float*)d_in[1];
  const float* gam   = (const float*)d_in[2];
  const float* bet   = (const float*)d_in[3];
  const float* Wf    = (const float*)d_in[4];
  const float* bias  = (const float*)d_in[5];
  const int*   tmask = (const int*)d_in[6];
  const int*   amask = (const int*)d_in[7];

  float* Tp  = (float*)d_out;                       // T_prime  [B,Lt,d]
  float* Aal = Tp + (size_t)BB * LT * DD;           // A_aligned[B,Lt,d]
  float* Sv  = Aal + (size_t)BB * LT * DD;          // S        [B,Lt]

  unsigned char* w = (unsigned char*)d_ws;
  size_t off = 0;
  auto take = [&](size_t bytes) -> void* {
    void* p = w + off;
    off += (bytes + 255) & ~(size_t)255;
    return p;
  };
  // bufA: LN(audio) bf16 -> A_norm bf16 -> A_prime^T bf16 (stream-ordered reuse)
  bf16*  bufA = (bf16*)take((size_t)BB * LA * DD * sizeof(bf16));
  bf16*  bufT = (bf16*)take((size_t)BB * LT * DD * sizeof(bf16));
  bf16*  Wbf  = (bf16*)take((size_t)DD * DD * sizeof(bf16));
  float* Ap   = (float*)take((size_t)BB * LA * DD * sizeof(float)); // A_prime
  float* lK   = (float*)take((size_t)BB * LT * LA * sizeof(float)); // log_K (134MB, L2-resident)
  float* lmu  = (float*)take((size_t)BB * LT * sizeof(float));
  float* lnu  = (float*)take((size_t)BB * LA * sizeof(float));
  float* lu   = (float*)take((size_t)BB * LT * sizeof(float));
  float* lv   = (float*)take((size_t)BB * LA * sizeof(float));

  // 1) W -> bf16; LayerNorm both streams -> bf16
  cast_bf16_kernel<<<1024, 256, 0, stream>>>(Wf, Wbf, DD * DD);
  ln_cast_kernel<<<BB * LT, 256, 0, stream>>>(text,  gam, bet, bufT);
  ln_cast_kernel<<<BB * LA, 256, 0, stream>>>(audio, gam, bet, bufA);

  // 2) projections (bf16 WMMA, fp32 accumulate, LDS-staged B)
  gemm_ln_proj_kernel<<<dim3(DD / 64, (BB * LT) / 128), 256, 0, stream>>>(
      bufT, Wbf, bias, Tp, DD, DD);
  gemm_ln_proj_kernel<<<dim3(DD / 64, (BB * LA) / 128), 256, 0, stream>>>(
      bufA, Wbf, bias, Ap, DD, DD);

  // 3) L2-normalize + bf16 cast (reuses the LN buffers)
  l2cast_kernel<<<BB * LT, 256, 0, stream>>>(Tp, bufT);
  l2cast_kernel<<<BB * LA, 256, 0, stream>>>(Ap, bufA);

  // 4) cost GEMM -> log_K = (cos - 1)/eps
  gemm_cost_kernel<<<dim3(LA / 64, LT / 128, BB), 256, 0, stream>>>(bufT, bufA, lK);

  // 5) A_prime -> transposed bf16 [b][d][La] for the final GEMM's B operand
  transpose_cast_kernel<<<dim3(DD / 32, LA / 32, BB), 256, 0, stream>>>(Ap, bufA);

  // 6) Sinkhorn marginals + init
  mask_log_kernel<<<BB, 256, 0, stream>>>(tmask, lmu, LT);
  mask_log_kernel<<<BB, 256, 0, stream>>>(amask, lnu, LA);
  zero_kernel<<<64, 256, 0, stream>>>(lu, BB * LT);
  zero_kernel<<<64, 256, 0, stream>>>(lv, BB * LA);

  // 7) 10 Sinkhorn iterations (u from v, then v from new u)
  for (int it = 0; it < SINK_ITERS; ++it) {
    sink_u_kernel<<<dim3(LT, BB), 256, 0, stream>>>(lK, lv, lmu, lu);
    sink_v_kernel<<<dim3(LA / 64, BB), dim3(64, 4), 0, stream>>>(lK, lu, lnu, lv);
  }

  // 8) A_aligned = exp(u + logK + v) @ A_prime, pi materialized in-register
  gemm_pi_align_kernel<<<dim3(DD / 64, LT / 128, BB), 256, 0, stream>>>(
      lK, lu, lv, bufA, Aal);

  // 9) S = rowwise dot(T_prime, A_aligned)
  s_dot_kernel<<<BB * LT, 256, 0, stream>>>(Tp, Aal, Sv);
}
